// BertSelfAttention_1726576855322
// MI455X (gfx1250) — compile-verified, matrix operand order checked by reading, not run on hardware
//
#include <hip/hip_runtime.h>
#include <hip/hip_bf16.h>

// ---------------------------------------------------------------------------
// BERT self-attention with relative_key_query position bias, MI455X (gfx1250).
// bf16 WMMA everywhere (v_wmma_f32_16x16x32_bf16), flash-style softmax (201MB
// score tensor never materialized). f32->bf16 conversion hoisted to a
// vectorized pre-pass; QKV GEMM is ping-pong double-buffered; the Toeplitz
// bias band is loaded once per 64-column tile and its Qd fragments are shared
// across all four column subtiles.
// ---------------------------------------------------------------------------

typedef __attribute__((ext_vector_type(16))) __bf16 v16bf;
typedef __attribute__((ext_vector_type(2)))  __bf16 v2bf;
typedef __attribute__((ext_vector_type(8)))  float  v8f;

union BF16Frag {
  v16bf v;
  uint4 q[2];
};

__device__ __forceinline__ unsigned int cvt2_bf16(float a, float b) {
  v2bf r = {(__bf16)a, (__bf16)b};   // -> v_cvt_pk_bf16_f32
  unsigned int u;
  __builtin_memcpy(&u, &r, 4);
  return u;
}

__device__ __forceinline__ v8f zero_v8f() {
  v8f z = {0.f, 0.f, 0.f, 0.f, 0.f, 0.f, 0.f, 0.f};
  return z;
}

// WMMA bf16 16x16x32 fragment element K(e) = kbase + (e/8)*16 + khalf*8 + e%8,
// row selected per lane by caller (M for A-frag, N for B-frag: both lane&15).
__device__ __forceinline__ v16bf load_frag_bf16(const __bf16* rowptr,
                                                int kbase, int khalf) {
  BF16Frag fr;
#pragma unroll
  for (int g = 0; g < 2; ++g) {
    fr.q[g] =
        *reinterpret_cast<const uint4*>(rowptr + kbase + g * 16 + khalf * 8);
  }
  return fr.v;
}

#define HS    768
#define S_LEN 1024
#define DH    64

// ---------------------------------------------------------------------------
// Pre-pass: vectorized f32 -> bf16 (4 elements per thread).
// ---------------------------------------------------------------------------
__global__ __launch_bounds__(256) void cvt_bf16_kernel(
    const float* __restrict__ src, __bf16* __restrict__ dst, int n4) {
  const int idx = blockIdx.x * blockDim.x + threadIdx.x;
  if (idx < n4) {
    float4 f = reinterpret_cast<const float4*>(src)[idx];
    uint2 p;
    p.x = cvt2_bf16(f.x, f.y);
    p.y = cvt2_bf16(f.z, f.w);
    reinterpret_cast<uint2*>(dst)[idx] = p;
  }
}

// ---------------------------------------------------------------------------
// Kernel 1: out[m,n] = sum_k X[m,k] * W[n,k] + bias[n]   (X @ W.T + b)
// Block = 256 threads (8 waves), each wave one 16x64 tile; grid (12, 32).
// K-loop ping-pong double-buffered so loads overlap WMMA.
// transposeV=1 stores per-head-transposed layout vt[b*768+n][s] for p@v.
// ---------------------------------------------------------------------------
__global__ __launch_bounds__(256) void qkv_proj_kernel(
    const __bf16* __restrict__ X, const __bf16* __restrict__ W,
    const float* __restrict__ bias, __bf16* __restrict__ out, int transposeV) {
  const int lane  = threadIdx.x & 31;
  const int wave  = threadIdx.x >> 5;
  const int khalf = lane >> 4;
  const int l16   = lane & 15;
  const int half8 = khalf * 8;
  const int m0    = blockIdx.y * 128 + wave * 16;
  const int n0    = blockIdx.x * 64;

  v8f acc[4];
#pragma unroll
  for (int t = 0; t < 4; ++t) acc[t] = zero_v8f();

  const __bf16* xrow = X + (size_t)(m0 + l16) * HS;
  const __bf16* wrow[4];
#pragma unroll
  for (int ns = 0; ns < 4; ++ns)
    wrow[ns] = W + (size_t)(n0 + ns * 16 + l16) * HS;

  v16bf aA, bA[4], aB, bB[4];
  aA = load_frag_bf16(xrow, 0, khalf);
#pragma unroll
  for (int ns = 0; ns < 4; ++ns) bA[ns] = load_frag_bf16(wrow[ns], 0, khalf);

#pragma unroll 1
  for (int kk = 0; kk < HS; kk += 64) {
    // prefetch the odd 32-slice while the even one computes
    aB = load_frag_bf16(xrow, kk + 32, khalf);
#pragma unroll
    for (int ns = 0; ns < 4; ++ns)
      bB[ns] = load_frag_bf16(wrow[ns], kk + 32, khalf);
#pragma unroll
    for (int ns = 0; ns < 4; ++ns)
      acc[ns] = __builtin_amdgcn_wmma_f32_16x16x32_bf16(
          false, aA, false, bA[ns], (short)0, acc[ns], false, false);
    // prefetch the next even 32-slice (clamped dummy on final iteration)
    const int kn = (kk + 64 < HS) ? kk + 64 : kk;
    aA = load_frag_bf16(xrow, kn, khalf);
#pragma unroll
    for (int ns = 0; ns < 4; ++ns) bA[ns] = load_frag_bf16(wrow[ns], kn, khalf);
#pragma unroll
    for (int ns = 0; ns < 4; ++ns)
      acc[ns] = __builtin_amdgcn_wmma_f32_16x16x32_bf16(
          false, aB, false, bB[ns], (short)0, acc[ns], false, false);
  }

#pragma unroll
  for (int ns = 0; ns < 4; ++ns) {
    const int n = n0 + ns * 16 + l16;
    const float bv = bias[n];
    if (!transposeV) {
#pragma unroll
      for (int i = 0; i < 8; ++i) {
        const int m = m0 + half8 + i;                 // C layout: vgpr i -> row
        out[(size_t)m * HS + n] = (__bf16)(acc[ns][i] + bv);
      }
    } else {
      const int bidx = m0 >> 10;                      // 128 | 1024: no crossing
      const int s0   = (m0 & 1023) + half8;
      uint4 pk;
      pk.x = cvt2_bf16(acc[ns][0] + bv, acc[ns][1] + bv);
      pk.y = cvt2_bf16(acc[ns][2] + bv, acc[ns][3] + bv);
      pk.z = cvt2_bf16(acc[ns][4] + bv, acc[ns][5] + bv);
      pk.w = cvt2_bf16(acc[ns][6] + bv, acc[ns][7] + bv);
      *reinterpret_cast<uint4*>(out + ((size_t)(bidx * HS + n)) * S_LEN + s0) = pk;
    }
  }
}

// ---------------------------------------------------------------------------
// Kernel 2: flash attention with Toeplitz relative-position bias.
// Grid (S/128, heads=12, B=4); 8 waves, each owns 16 query rows, streams
// key/value columns 64 at a time with online softmax. The dist_emb band
// (80 rows) is loaded once per column tile; Qd fragments are shared across
// the four 16-column subtiles (t = js - cs takes only 5 values).
// ---------------------------------------------------------------------------
__global__ __launch_bounds__(256) void attn_kernel(
    const __bf16* __restrict__ qbuf, const __bf16* __restrict__ kbuf,
    const __bf16* __restrict__ vt, const __bf16* __restrict__ distb,
    const float* __restrict__ mask, float* __restrict__ out) {
  __shared__ float s_qd[8][16][81];                     // Qd[l, J] 80-wide band
  __shared__ float s_kd[8][16][33];                     // Kd[r, j] 31-wide band
  __shared__ __align__(16) __bf16 s_p[8][16][64];       // P accum -> A-frag relayout

  const int lane  = threadIdx.x & 31;
  const int wave  = threadIdx.x >> 5;
  const int khalf = lane >> 4;
  const int l16   = lane & 15;
  const int half8 = khalf * 8;
  const int l0    = blockIdx.x * 128 + wave * 16;
  const int h     = blockIdx.y;
  const int b     = blockIdx.z;

  const __bf16* qbase = qbuf + (size_t)(b * S_LEN) * HS + h * DH;
  const __bf16* kbase = kbuf + (size_t)(b * S_LEN) * HS + h * DH;
  const __bf16* vbase = vt + (size_t)(b * HS + h * DH) * S_LEN;
  const float*  mrow  = mask + (size_t)b * S_LEN;

  // Q A-fragments: reused for q@k^T and q@dist^T across all column tiles.
  v16bf aq[2];
#pragma unroll
  for (int ks = 0; ks < 2; ++ks)
    aq[ks] = load_frag_bf16(qbase + (size_t)(l0 + l16) * HS, ks * 32, khalf);

  v8f o[4];
  float mrun[8], lrun[8];
#pragma unroll
  for (int t = 0; t < 4; ++t) o[t] = zero_v8f();
#pragma unroll
  for (int i = 0; i < 8; ++i) { mrun[i] = -3.0e38f; lrun[i] = 0.f; }

#pragma unroll 1
  for (int r0 = 0; r0 < S_LEN; r0 += 64) {
    if (r0 + 64 < S_LEN)
      __builtin_prefetch(kbase + (size_t)(r0 + 64 + l16) * HS, 0, 3);

    // ---- K fragments, batched (same lane data is B-frag for q@k^T and
    //      A-frag for Kd thanks to the symmetric fragment lane mapping) ----
    v16bf kfrag[4][2];
#pragma unroll
    for (int cs = 0; cs < 4; ++cs) {
      const __bf16* krow = kbase + (size_t)(r0 + cs * 16 + l16) * HS;
#pragma unroll
      for (int ks = 0; ks < 2; ++ks)
        kfrag[cs][ks] = load_frag_bf16(krow, ks * 32, khalf);
    }

    // ---- scores: q @ k^T ----
    v8f sc[4];
#pragma unroll
    for (int cs = 0; cs < 4; ++cs) {
      sc[cs] = zero_v8f();
#pragma unroll
      for (int ks = 0; ks < 2; ++ks)
        sc[cs] = __builtin_amdgcn_wmma_f32_16x16x32_bf16(
            false, aq[ks], false, kfrag[cs][ks], (short)0, sc[cs], false, false);
    }

    // ---- dist_emb band: rows base0-48 .. base0+31 (5 x 16 rows) ----
    const int base0 = l0 - r0 + 1008;
    v16bf db[5][2];
#pragma unroll
    for (int tt = 0; tt < 5; ++tt) {
      int row = base0 - 48 + tt * 16 + l16;
      row = row < 0 ? 0 : (row > 2046 ? 2046 : row);   // clamp unused edges
#pragma unroll
      for (int ks = 0; ks < 2; ++ks)
        db[tt][ks] = load_frag_bf16(distb + (size_t)row * DH, ks * 32, khalf);
    }

    // ---- Qd: 5 shared fragments, stored to LDS band ----
#pragma unroll
    for (int tt = 0; tt < 5; ++tt) {
      v8f qd = zero_v8f();
#pragma unroll
      for (int ks = 0; ks < 2; ++ks)
        qd = __builtin_amdgcn_wmma_f32_16x16x32_bf16(
            false, aq[ks], false, db[tt][ks], (short)0, qd, false, false);
#pragma unroll
      for (int i = 0; i < 8; ++i)
        s_qd[wave][half8 + i][tt * 16 + l16] = qd[i];
    }
    asm volatile("s_wait_dscnt 0" ::: "memory");

    // ---- Kd per column subtile + diagonal gather ----
#pragma unroll
    for (int cs = 0; cs < 4; ++cs) {
#pragma unroll
      for (int js = 0; js < 2; ++js) {
        const int tt = js - cs + 3;                    // in [0, 4]
        v8f kd = zero_v8f();
#pragma unroll
        for (int ks = 0; ks < 2; ++ks)
          kd = __builtin_amdgcn_wmma_f32_16x16x32_bf16(
              false, kfrag[cs][ks], false, db[tt][ks], (short)0, kd, false, false);
#pragma unroll
        for (int i = 0; i < 8; ++i)
          s_kd[wave][half8 + i][js * 16 + l16] = kd[i];
      }
      asm volatile("s_wait_dscnt 0" ::: "memory");
#pragma unroll
      for (int i = 0; i < 8; ++i) {
        const int li = half8 + i;
        sc[cs][i] += s_qd[wave][li][li - l16 + 63 - cs * 16] +
                     s_kd[wave][l16][li - l16 + 15];
      }
      asm volatile("s_wait_dscnt 0" ::: "memory");
    }

    // ---- scale 1/sqrt(64), add mask, online softmax ----
#pragma unroll
    for (int cs = 0; cs < 4; ++cs) {
      const float mv = mrow[r0 + cs * 16 + l16];
#pragma unroll
      for (int i = 0; i < 8; ++i) sc[cs][i] = sc[cs][i] * 0.125f + mv;
    }
    float rmax[8];
#pragma unroll
    for (int i = 0; i < 8; ++i) {
      float v = fmaxf(fmaxf(sc[0][i], sc[1][i]), fmaxf(sc[2][i], sc[3][i]));
#pragma unroll
      for (int off = 1; off < 16; off <<= 1) v = fmaxf(v, __shfl_xor(v, off, 32));
      rmax[i] = v;
    }
    float alpha[8];
#pragma unroll
    for (int i = 0; i < 8; ++i) {
      const float mnew = fmaxf(mrun[i], rmax[i]);
      alpha[i] = __expf(mrun[i] - mnew);
      mrun[i] = mnew;
    }
    float rsum[8];
#pragma unroll
    for (int i = 0; i < 8; ++i) rsum[i] = 0.f;
#pragma unroll
    for (int cs = 0; cs < 4; ++cs)
#pragma unroll
      for (int i = 0; i < 8; ++i) {
        const float p = __expf(sc[cs][i] - mrun[i]);
        sc[cs][i] = p;
        rsum[i] += p;
      }
#pragma unroll
    for (int i = 0; i < 8; ++i) {
      float v = rsum[i];
#pragma unroll
      for (int off = 1; off < 16; off <<= 1) v += __shfl_xor(v, off, 32);
      lrun[i] = lrun[i] * alpha[i] + v;
    }
#pragma unroll
    for (int t = 0; t < 4; ++t)
#pragma unroll
      for (int i = 0; i < 8; ++i) o[t][i] *= alpha[i];

    // ---- P (C-layout) -> LDS -> A-fragments; V^T B-frags batched ----
#pragma unroll
    for (int cs = 0; cs < 4; ++cs)
#pragma unroll
      for (int i = 0; i < 8; ++i)
        s_p[wave][half8 + i][cs * 16 + l16] = (__bf16)sc[cs][i];
    asm volatile("s_wait_dscnt 0" ::: "memory");

    v16bf ap[2];
#pragma unroll
    for (int ks = 0; ks < 2; ++ks)
      ap[ks] = load_frag_bf16(&s_p[wave][l16][0], ks * 32, khalf);

    v16bf bvf[4][2];
#pragma unroll
    for (int ds = 0; ds < 4; ++ds) {
      const __bf16* vrow = vbase + (size_t)(ds * 16 + l16) * S_LEN + r0;
#pragma unroll
      for (int ks = 0; ks < 2; ++ks)
        bvf[ds][ks] = load_frag_bf16(vrow, ks * 32, khalf);
    }
#pragma unroll
    for (int ds = 0; ds < 4; ++ds)
#pragma unroll
      for (int ks = 0; ks < 2; ++ks)
        o[ds] = __builtin_amdgcn_wmma_f32_16x16x32_bf16(
            false, ap[ks], false, bvf[ds][ks], (short)0, o[ds], false, false);
    asm volatile("s_wait_dscnt 0" ::: "memory");
  }

  float* orow = out + (size_t)(b * S_LEN + l0) * HS + h * DH;
#pragma unroll
  for (int i = 0; i < 8; ++i) {
    const float inv = 1.f / lrun[i];
    const int li = half8 + i;
#pragma unroll
    for (int ds = 0; ds < 4; ++ds)
      orow[(size_t)li * HS + ds * 16 + l16] = o[ds][i] * inv;
  }
}

// ---------------------------------------------------------------------------
extern "C" void kernel_launch(void* const* d_in, const int* in_sizes, int n_in,
                              void* d_out, int out_size, void* d_ws, size_t ws_size,
                              hipStream_t stream) {
  const float* X    = (const float*)d_in[0];  // hidden_states [4,1024,768]
  const float* mask = (const float*)d_in[1];  // [4,1,1,1024]
  const float* Wq   = (const float*)d_in[2];
  const float* bq   = (const float*)d_in[3];
  const float* Wk   = (const float*)d_in[4];
  const float* bk   = (const float*)d_in[5];
  const float* Wv   = (const float*)d_in[6];
  const float* bv   = (const float*)d_in[7];
  const float* de   = (const float*)d_in[8];  // dist_emb [2047,64]

  const size_t NX = (size_t)4096 * 768;   // hidden / q / k / v element count
  const size_t NW = (size_t)768 * 768;    // weight element count
  const size_t ND = (size_t)2047 * 64;    // dist_emb element count

  // Workspace (bf16): q,k row-major; v per-head-transposed [4*768,1024];
  // pre-converted X, Wq, Wk, Wv, dist_emb.
  __bf16* qbuf = (__bf16*)d_ws;
  __bf16* kbuf = qbuf + NX;
  __bf16* vtb  = kbuf + NX;
  __bf16* xbf  = vtb + NX;
  __bf16* wqb  = xbf + NX;
  __bf16* wkb  = wqb + NW;
  __bf16* wvb  = wkb + NW;
  __bf16* dbf  = wvb + NW;

  dim3 blk(256, 1, 1);

  // f32 -> bf16 pre-pass (vector of 4 per thread)
  cvt_bf16_kernel<<<dim3((unsigned)(NX / 4 + 255) / 256), blk, 0, stream>>>(X, xbf, (int)(NX / 4));
  cvt_bf16_kernel<<<dim3((unsigned)(NW / 4 + 255) / 256), blk, 0, stream>>>(Wq, wqb, (int)(NW / 4));
  cvt_bf16_kernel<<<dim3((unsigned)(NW / 4 + 255) / 256), blk, 0, stream>>>(Wk, wkb, (int)(NW / 4));
  cvt_bf16_kernel<<<dim3((unsigned)(NW / 4 + 255) / 256), blk, 0, stream>>>(Wv, wvb, (int)(NW / 4));
  cvt_bf16_kernel<<<dim3((unsigned)(ND / 4 + 255) / 256), blk, 0, stream>>>(de, dbf, (int)(ND / 4));

  dim3 gproj(12, 32, 1);  // N/64, M/128
  qkv_proj_kernel<<<gproj, blk, 0, stream>>>(xbf, wqb, bq, qbuf, 0);
  qkv_proj_kernel<<<gproj, blk, 0, stream>>>(xbf, wkb, bk, kbuf, 0);
  qkv_proj_kernel<<<gproj, blk, 0, stream>>>(xbf, wvb, bv, vtb, 1);

  dim3 gattn(8, 12, 4);   // S/128, heads, batch
  attn_kernel<<<gattn, blk, 0, stream>>>(qbuf, kbuf, vtb, dbf, mask, (float*)d_out);
}
